// BasicBlock_46626164966153
// MI455X (gfx1250) — matrix operand order repaired and checked by reading, not compiled, not run on hardware
//
#include <hip/hip_runtime.h>

// ---------------- CDNA5 types ----------------
typedef __attribute__((ext_vector_type(16))) __bf16        v16bf;
typedef __attribute__((ext_vector_type(8)))  float         v8f;
typedef __attribute__((ext_vector_type(4)))  unsigned int  u32x4;
typedef __attribute__((ext_vector_type(8)))  int           i32x8;
typedef __attribute__((ext_vector_type(4)))  int           i32x4;

union ABFrag { uint4 q[2]; v16bf v; };

// ---------------- helpers ----------------
__device__ __forceinline__ unsigned short f2bf(float f) {
    unsigned u = __float_as_uint(f);
    u += 0x7fffu + ((u >> 16) & 1u);   // round-to-nearest-even
    return (unsigned short)(u >> 16);
}
__device__ __forceinline__ float bf2f(unsigned short s) {
    return __uint_as_float(((unsigned)s) << 16);
}

// Problem constants
#define BB 32
#define CC 128
#define HH 56
#define WW 56
#define HW (HH*WW)          // 3136
#define K9 9
#define CNT ((float)(BB*HH*WW))   // BN count = 100352

// ---------------- TDM 3D tile load (D# per ISA 8.3-8.5) ----------------
__device__ __forceinline__ void tdm_load_3d(
        unsigned lds_addr, const void* gaddr, int dsize_code,
        unsigned td0, unsigned td1, unsigned td2,          // tensor dims (OOB -> 0)
        unsigned tile0, unsigned tile1, unsigned tile2,    // tile dims (x,y,z)
        unsigned long long str0, unsigned long long str1)  // y-step, z-step (elements)
{
    unsigned long long ga = (unsigned long long)(size_t)gaddr;
    u32x4 g0;
    g0.x = 1u;                                             // count=1 valid descriptor
    g0.y = lds_addr;                                       // LDS byte offset
    g0.z = (unsigned)(ga & 0xffffffffull);
    g0.w = ((unsigned)(ga >> 32) & 0x01ffffffu) | 0x80000000u;  // addr[56:32] | type=2
    i32x8 g1 = {0,0,0,0,0,0,0,0};
    g1[0] = dsize_code << 16;                              // data_size
    g1[1] = (int)((td0 & 0xffffu) << 16);
    g1[2] = (int)(((td0 >> 16) & 0xffffu) | ((td1 & 0xffffu) << 16));
    g1[3] = (int)(((td1 >> 16) & 0xffffu) | (tile0 << 16));
    g1[4] = (int)(tile1 | (tile2 << 16));
    g1[5] = (int)(str0 & 0xffffffffull);
    g1[6] = (int)(((str0 >> 32) & 0xffffu) | ((str1 & 0xffffu) << 16));
    g1[7] = (int)((str1 >> 16) & 0xffffffffull);
    i32x4 g2 = {(int)td2, 0, 0, 0};
    i32x4 g3 = {0, 0, 0, 0};
#if defined(__clang_major__) && (__clang_major__ >= 23)
    i32x8 g4 = {0,0,0,0,0,0,0,0};
    __builtin_amdgcn_tensor_load_to_lds(g0, g1, g2, g3, g4, 0);
#else
    __builtin_amdgcn_tensor_load_to_lds(g0, g1, g2, g3, 0);
#endif
}

// ---------------- weight prepack: [o][ci][3][3] f32 -> [k9][o][ci] bf16 --------
__global__ void prepack_kernel(const float* __restrict__ w1,
                               const float* __restrict__ w2,
                               unsigned short* __restrict__ p1,
                               unsigned short* __restrict__ p2) {
    int i = blockIdx.x * 256 + threadIdx.x;
    const int N = K9 * CC * CC;        // 147456
    if (i >= 2 * N) return;
    const float* src = (i < N) ? w1 : w2;
    unsigned short* dst = (i < N) ? p1 : p2;
    int j = (i < N) ? i : i - N;
    int k9 = j / (CC * CC);
    int rem = j - k9 * (CC * CC);
    int o  = rem >> 7;
    int ci = rem & 127;
    dst[j] = f2bf(src[(o * CC + ci) * K9 + k9]);
}

// ---------------- shared WMMA core: 16 out-ch x 64 px per wave ----------------
// Xs: [row 6][col 18][ci 128] bf16. 9 taps x 4 ci-chunks -> 144 WMMA / wave.
__device__ __forceinline__ void conv_core(
        const unsigned short* __restrict__ wpk,
        const unsigned short* Xs,
        int obase, int lmod, int lhalf, v8f acc[4])
{
    const int kb = lhalf * 16;
    const unsigned short* wlane = wpk + (obase + lmod) * CC + kb;
    auto loadA = [&](int idx36) -> ABFrag {
        const int k9 = idx36 >> 2, cc = idx36 & 3;
        ABFrag a;
        const uint4* wp = (const uint4*)(wlane + k9 * CC * CC + cc * 32);
        a.q[0] = wp[0];     // K = kb .. kb+7
        a.q[1] = wp[1];     // K = kb+8 .. kb+15
        return a;
    };
    ABFrag aCur = loadA(0);
    #pragma unroll 1
    for (int k9 = 0; k9 < K9; ++k9) {
        const int kh = k9 / 3;
        const int kw = k9 - 3 * kh;
        if (k9 + 1 < K9)
            __builtin_prefetch(wpk + (k9 + 1) * CC * CC + (obase + lmod) * CC, 0, 1);
        const unsigned short* xb = &Xs[(kh * 18 + lmod + kw) * CC + lhalf * 16];
        #pragma unroll
        for (int cc = 0; cc < 4; ++cc) {
            ABFrag aNxt = aCur;                 // software pipeline next A fragment
            const int ni = k9 * 4 + cc + 1;
            if (ni < 36) aNxt = loadA(ni);
            ABFrag bfr[4];                      // preload 4 B fragments (ds pipelines)
            #pragma unroll
            for (int t = 0; t < 4; ++t) {
                const uint4* bp = (const uint4*)(xb + t * 18 * CC + cc * 32);
                bfr[t].q[0] = bp[0];
                bfr[t].q[1] = bp[1];
            }
            #pragma unroll
            for (int t = 0; t < 4; ++t)
                acc[t] = __builtin_amdgcn_wmma_f32_16x16x32_bf16(
                    false, aCur.v, false, bfr[t].v, (short)0, acc[t], false, false);
            aCur = aNxt;
        }
    }
}

// -------- shared epilogue: NHWC bf16 packed store + shuffle-reduced BN stats -----
__device__ __forceinline__ void store_nhwc_stats(
        unsigned short* __restrict__ out,     // [B][H][W][C] bf16
        const v8f acc[4], int b, int h0, int wcol, bool valid,
        int obase, int lmod, int lhalf,
        float* sSum, float* sSq)
{
    float sv[8], sv2[8];
    #pragma unroll
    for (int r = 0; r < 8; ++r) { sv[r] = 0.f; sv2[r] = 0.f; }

    #pragma unroll
    for (int t = 0; t < 4; ++t) {
        const int h = h0 + t;
        if (valid) {
            uint4 pk;
            unsigned* pw = (unsigned*)&pk;
            #pragma unroll
            for (int r = 0; r < 4; ++r)
                pw[r] = (unsigned)f2bf(acc[t][2 * r]) |
                        ((unsigned)f2bf(acc[t][2 * r + 1]) << 16);
            *(uint4*)&out[(((size_t)b * HH + h) * WW + wcol) * CC + obase + 8 * lhalf] = pk;
        }
        #pragma unroll
        for (int r = 0; r < 8; ++r) {          // in-lane pre-reduction over rows
            float v = valid ? acc[t][r] : 0.f; // masked cols contribute exact 0
            sv[r]  += v;
            sv2[r] += v * v;
        }
    }
    // one 16-lane butterfly per channel (masks stay within each lane half)
    #pragma unroll
    for (int r = 0; r < 8; ++r) {
        float v = sv[r], v2 = sv2[r];
        #pragma unroll
        for (int m = 8; m > 0; m >>= 1) {
            v  += __shfl_xor(v,  m, 32);
            v2 += __shfl_xor(v2, m, 32);
        }
        if (lmod == 0) {
            const int o = obase + r + 8 * lhalf;
            atomicAdd(&sSum[o], v);
            atomicAdd(&sSq[o],  v2);
        }
    }
}

// ---------------- conv1: x f32 NCHW -> y1 bf16 NHWC (+ BN1 stats) ----------------
__global__ __launch_bounds__(256) void conv1_kernel(
        const float* __restrict__ x,
        const unsigned short* __restrict__ wpk,
        unsigned short* __restrict__ y1,           // [B][H][W][C] bf16
        float* __restrict__ gsum, float* __restrict__ gsq)
{
    __shared__ __align__(16) unsigned short Xs[6 * 18 * CC];   // [row][col][ci]
    __shared__ __align__(16) float          Xtdm[6 * 18 * CC]; // TDM landing [ci][row][col]
    __shared__ float sSum[CC];
    __shared__ float sSq[CC];

    const int b  = blockIdx.z;
    const int h0 = blockIdx.y * 4;
    const int w0 = blockIdx.x * 16;
    const int wid = threadIdx.x >> 5;

    if (threadIdx.x < CC) { sSum[threadIdx.x] = 0.f; sSq[threadIdx.x] = 0.f; }

    const bool interior = (h0 > 0) && (w0 > 0);
    if (interior) {
        if (wid == 0) {
            const float* gp = x + ((size_t)b * CC * HH + (h0 - 1)) * WW + (w0 - 1);
            tdm_load_3d((unsigned)(size_t)(void*)Xtdm, gp, /*4B*/2,
                        (unsigned)(WW - (w0 - 1)), (unsigned)(HH - (h0 - 1)), CC,
                        18, 6, CC, WW, HW);
            __builtin_amdgcn_s_wait_tensorcnt(0);
        }
        __syncthreads();
        // transpose+convert: [ci][row][col] f32 -> [row][col][ci] bf16
        for (int e = threadIdx.x; e < CC * 108; e += 256) {
            int ci  = e / 108;
            int rem = e - ci * 108;
            int sr  = rem / 18;
            int sc  = rem - sr * 18;
            Xs[(sr * 18 + sc) * CC + ci] = f2bf(Xtdm[e]);
        }
    } else {
        for (int e = threadIdx.x; e < CC * 108; e += 256) {
            int ci  = e / 108;
            int rem = e - ci * 108;
            int sr  = rem / 18;
            int sc  = rem - sr * 18;
            int h = h0 + sr - 1, w = w0 + sc - 1;
            float v = 0.f;
            if (h >= 0 && h < HH && w >= 0 && w < WW)
                v = x[((b * CC + ci) * HH + h) * WW + w];
            Xs[(sr * 18 + sc) * CC + ci] = f2bf(v);
        }
    }
    __syncthreads();

    const int lane  = threadIdx.x & 31;
    const int lmod  = lane & 15;
    const int lhalf = lane >> 4;
    const int obase = wid * 16;

    v8f acc[4];
    #pragma unroll
    for (int t = 0; t < 4; ++t)
        #pragma unroll
        for (int i = 0; i < 8; ++i) acc[t][i] = 0.f;

    conv_core(wpk, Xs, obase, lmod, lhalf, acc);

    const int wcol = w0 + lmod;
    store_nhwc_stats(y1, acc, b, h0, wcol, wcol < WW, obase, lmod, lhalf, sSum, sSq);

    __syncthreads();
    if (threadIdx.x < CC) {
        atomicAdd(&gsum[threadIdx.x], sSum[threadIdx.x]);
        atomicAdd(&gsq[threadIdx.x],  sSq[threadIdx.x]);
    }
}

// -------- conv2: y1 bf16 NHWC (BN1+ReLU on the fly) -> y2 bf16 NHWC (+ BN2 stats) ----
__global__ __launch_bounds__(256) void conv2_kernel(
        const unsigned short* __restrict__ y1,     // [B][H][W][C] bf16
        const unsigned short* __restrict__ wpk,
        const float* __restrict__ bscale, const float* __restrict__ bbias,
        unsigned short* __restrict__ y2,           // [B][H][W][C] bf16
        float* __restrict__ gsum, float* __restrict__ gsq)
{
    __shared__ __align__(16) unsigned short Xs[6 * 18 * CC];   // TDM lands here directly
    __shared__ float sSum[CC];
    __shared__ float sSq[CC];

    const int b  = blockIdx.z;
    const int h0 = blockIdx.y * 4;
    const int w0 = blockIdx.x * 16;
    const int wid = threadIdx.x >> 5;

    if (threadIdx.x < CC) { sSum[threadIdx.x] = 0.f; sSq[threadIdx.x] = 0.f; }

    const bool interior = (h0 > 0) && (w0 > 0);
    if (interior) {
        if (wid == 0) {
            // NHWC: x=ci(contig,128), y=col(18, stride C), z=row(6, stride W*C)
            const unsigned short* gp = y1 +
                (((size_t)b * HH + (h0 - 1)) * WW + (w0 - 1)) * CC;
            tdm_load_3d((unsigned)(size_t)(void*)Xs, gp, /*2B*/1,
                        CC, (unsigned)(WW - (w0 - 1)), (unsigned)(HH - (h0 - 1)),
                        CC, 18, 6, CC, (unsigned long long)WW * CC);
            __builtin_amdgcn_s_wait_tensorcnt(0);
        }
        __syncthreads();
        // in-place BN+ReLU; zero where (h,w) outside image (pad AFTER activation)
        for (int e = threadIdx.x; e < CC * 108; e += 256) {
            int ci  = e & 127;
            int pix = e >> 7;
            int sr  = pix / 18;
            int sc  = pix - sr * 18;
            int h = h0 + sr - 1, w = w0 + sc - 1;   // >= 0 since interior
            float v = 0.f;
            if (h < HH && w < WW)
                v = fmaxf(bf2f(Xs[e]) * bscale[ci] + bbias[ci], 0.f);
            Xs[e] = f2bf(v);
        }
    } else {
        for (int e = threadIdx.x; e < CC * 108; e += 256) {
            int ci  = e & 127;
            int pix = e >> 7;
            int sr  = pix / 18;
            int sc  = pix - sr * 18;
            int h = h0 + sr - 1, w = w0 + sc - 1;
            float v = 0.f;
            if (h >= 0 && h < HH && w >= 0 && w < WW) {
                v = bf2f(y1[(((size_t)b * HH + h) * WW + w) * CC + ci]);
                v = fmaxf(v * bscale[ci] + bbias[ci], 0.f);
            }
            Xs[e] = f2bf(v);
        }
    }
    __syncthreads();

    const int lane  = threadIdx.x & 31;
    const int lmod  = lane & 15;
    const int lhalf = lane >> 4;
    const int obase = wid * 16;

    v8f acc[4];
    #pragma unroll
    for (int t = 0; t < 4; ++t)
        #pragma unroll
        for (int i = 0; i < 8; ++i) acc[t][i] = 0.f;

    conv_core(wpk, Xs, obase, lmod, lhalf, acc);

    const int wcol = w0 + lmod;
    store_nhwc_stats(y2, acc, b, h0, wcol, wcol < WW, obase, lmod, lhalf, sSum, sSq);

    __syncthreads();
    if (threadIdx.x < CC) {
        atomicAdd(&gsum[threadIdx.x], sSum[threadIdx.x]);
        atomicAdd(&gsq[threadIdx.x],  sSq[threadIdx.x]);
    }
}

// ---------------- BN stats -> affine (scale, bias) ----------------
__global__ void bnstats_kernel(const float* __restrict__ sum,
                               const float* __restrict__ sq,
                               const float* __restrict__ gamma,
                               const float* __restrict__ beta,
                               float* __restrict__ scale,
                               float* __restrict__ bias) {
    int c = threadIdx.x;
    float m   = sum[c] / CNT;
    float var = sq[c] / CNT - m * m;
    float inv = rsqrtf(var + 1e-5f);
    float s   = gamma[c] * inv;
    scale[c] = s;
    bias[c]  = beta[c] - m * s;
}

// ---------------- epilogue: relu(bn2(y2) + x), y2 bf16 NHWC (L2-resident) ----------------
__global__ __launch_bounds__(256) void epilogue_kernel(
        const unsigned short* __restrict__ y2,     // [B][H][W][C] bf16
        const float* __restrict__ x,               // [B][C][H][W] f32
        const float* __restrict__ scale2, const float* __restrict__ bias2,
        float* __restrict__ out, int n4) {
    int i = blockIdx.x * 256 + threadIdx.x;
    if (i >= n4) return;
    int base = i * 4;                    // flat NCHW element index
    int b    = base / (CC * HW);
    int rem  = base - b * (CC * HW);
    int c    = rem / HW;
    int p    = rem - c * HW;             // pixel index, p..p+3 in same (b,c)
    const unsigned short* yp = y2 + ((size_t)b * HW + p) * CC + c;
    float s = scale2[c], bb = bias2[c];
    float4 xi = ((const float4*)x)[i];
    float4 o;
    o.x = fmaxf(bf2f(yp[0 * CC]) * s + bb + xi.x, 0.f);
    o.y = fmaxf(bf2f(yp[1 * CC]) * s + bb + xi.y, 0.f);
    o.z = fmaxf(bf2f(yp[2 * CC]) * s + bb + xi.z, 0.f);
    o.w = fmaxf(bf2f(yp[3 * CC]) * s + bb + xi.w, 0.f);
    ((float4*)out)[i] = o;
}

// ---------------- launch ----------------
extern "C" void kernel_launch(void* const* d_in, const int* in_sizes, int n_in,
                              void* d_out, int out_size, void* d_ws, size_t ws_size,
                              hipStream_t stream) {
    const float* x  = (const float*)d_in[0];
    const float* w1 = (const float*)d_in[1];
    const float* g1 = (const float*)d_in[2];
    const float* b1 = (const float*)d_in[3];
    const float* w2 = (const float*)d_in[4];
    const float* g2 = (const float*)d_in[5];
    const float* b2 = (const float*)d_in[6];

    char* ws = (char*)d_ws;
    const size_t WPK_BYTES = (size_t)K9 * CC * CC * 2;          // 294912
    unsigned short* wpk1 = (unsigned short*)(ws);
    unsigned short* wpk2 = (unsigned short*)(ws + WPK_BYTES);
    float* stats  = (float*)(ws + 2 * WPK_BYTES);               // 8 x 128 floats
    float* sum1   = stats;        float* sq1   = stats + 128;
    float* sum2   = stats + 256;  float* sq2   = stats + 384;
    float* scale1 = stats + 512;  float* bias1 = stats + 640;
    float* scale2 = stats + 768;  float* bias2 = stats + 896;
    const size_t Y1_OFF = 2 * WPK_BYTES + 4096;                 // 593920
    unsigned short* y1 = (unsigned short*)(ws + Y1_OFF);        // bf16 NHWC, 25.7 MB
    const size_t Y2_OFF = Y1_OFF + (size_t)BB * CC * HW * 2;
    unsigned short* y2 = (unsigned short*)(ws + Y2_OFF);        // bf16 NHWC, 25.7 MB

    hipMemsetAsync(stats, 0, 4 * 128 * sizeof(float), stream);

    prepack_kernel<<<(2 * K9 * CC * CC + 255) / 256, 256, 0, stream>>>(w1, w2, wpk1, wpk2);

    dim3 cgrid(WW / 16 + 1, HH / 4, BB);   // (4, 14, 32)
    conv1_kernel<<<cgrid, 256, 0, stream>>>(x, wpk1, y1, sum1, sq1);
    bnstats_kernel<<<1, 128, 0, stream>>>(sum1, sq1, g1, b1, scale1, bias1);
    conv2_kernel<<<cgrid, 256, 0, stream>>>(y1, wpk2, scale1, bias1, y2, sum2, sq2);
    bnstats_kernel<<<1, 128, 0, stream>>>(sum2, sq2, g2, b2, scale2, bias2);

    const int n4 = BB * CC * HW / 4;       // 3211264
    epilogue_kernel<<<n4 / 256, 256, 0, stream>>>(y2, x, scale2, bias2, (float*)d_out, n4);
}
